// GINLayer_37065567764984
// MI455X (gfx1250) — matrix operand order repaired and checked by reading, not compiled, not run on hardware
//
#include <hip/hip_runtime.h>

// GIN layer for MI455X (gfx1250, wave32).
//  - scatter-add message passing via global_atomic_add_f32 (L2-resident)
//  - both 64x64 GEMMs via v_wmma_f32_16x16x32_f16 (f16 A/B, f32 accum)
//  - BatchNorm stats via wave-local sums -> LDS ds_add_f32 -> global atomics

typedef __attribute__((ext_vector_type(16))) _Float16 v16h;
typedef __attribute__((ext_vector_type(8)))  _Float16 h8;
typedef __attribute__((ext_vector_type(8)))  float    v8f;
typedef __attribute__((ext_vector_type(4)))  float    v4f;

#define DIMF 64
#define BN_EPS 1e-5f

// ---------------- K0: zero neigh accumulator + BN stat accumulators ----------
__global__ void k_zero(float* __restrict__ neigh, float* __restrict__ stats,
                       int nvec4) {
  int i = blockIdx.x * blockDim.x + threadIdx.x;
  int stride = gridDim.x * blockDim.x;
  v4f z = {0.f, 0.f, 0.f, 0.f};
  for (int j = i; j < nvec4; j += stride) ((v4f*)neigh)[j] = z;
  if (i < 128) stats[i] = 0.f;
}

// ---------------- K1: edge scatter-add: neigh[dst] += h[src] ----------------
// One wave per edge; each lane handles a float2 (2 channels) -> coalesced
// 256B row gather + 2 fire-and-forget f32 atomics per lane.
__global__ __launch_bounds__(256) void k_scatter(const float* __restrict__ h,
                                                 const int* __restrict__ src,
                                                 const int* __restrict__ dst,
                                                 float* __restrict__ neigh,
                                                 int nEdges) {
  int gtid = blockIdx.x * blockDim.x + threadIdx.x;
  int e    = gtid >> 5;
  int lane = gtid & 31;
  if (e >= nEdges) return;
  int s = src[e];
  int d = dst[e];
  float2 v = ((const float2*)(h + (size_t)s * DIMF))[lane];
  float* np = neigh + (size_t)d * DIMF + lane * 2;
  atomicAdd(np,     v.x);
  atomicAdd(np + 1, v.y);
}

// ---------------- K2: zh = (f16)(h + neigh)  (eps_gin == 0) -----------------
__global__ void k_zh(const float* __restrict__ h, const float* __restrict__ neigh,
                     _Float16* __restrict__ zh, int n) {
  int i = blockIdx.x * blockDim.x + threadIdx.x;
  if (i < n) zh[i] = (_Float16)(h[i] + neigh[i]);
}

// ---------------- Kw: pack W1/W2 into B-fragment blocked f16 layout ---------
// Wh[(kblk*64 + col)*16 + e] = W[kblk*16 + e][col]  so each lane's v16h B
// fragment (lane group g, step s -> kblk = 2s+g) is one contiguous 32B load.
__global__ void k_packW(const float* __restrict__ W1, const float* __restrict__ W2,
                        _Float16* __restrict__ Wh1, _Float16* __restrict__ Wh2) {
  int i = blockIdx.x * blockDim.x + threadIdx.x;   // 0..4095
  if (i >= DIMF * DIMF) return;
  int k = i >> 6, c = i & 63;
  int kblk = k >> 4, e = k & 15;
  int di = ((kblk * 64 + c) << 4) + e;
  Wh1[di] = (_Float16)W1[i];
  Wh2[di] = (_Float16)W2[i];
}

// A fragment for v_wmma_f32_16x16x32_f16, contiguous k order (same as packed B):
// lane (g = lane>>4, m = lane&15), K-step s covers K = 32s..32s+31.
// elements 0..7  -> K = 32s + 16g + 0..7
// elements 8..15 -> K = 32s + 16g + 8 + 0..7
__device__ inline v16h load_A(const _Float16* __restrict__ zh, int row, int g, int s) {
  const _Float16* base = zh + (size_t)row * DIMF + s * 32 + g * 16;
  h8 r1 = *(const h8*)(base);
  h8 r2 = *(const h8*)(base + 8);
  v16h a;
#pragma unroll
  for (int e = 0; e < 8; ++e) { a[e] = r1[e]; a[8 + e] = r2[e]; }
  return a;
}

__device__ inline void gemm_strip(const _Float16* __restrict__ Ah,
                                  const _Float16* __restrict__ Wh,
                                  int R, int g, int mn, v8f acc[4]) {
#pragma unroll
  for (int s = 0; s < 2; ++s) {
    v16h a = load_A(Ah, R + mn, g, s);
#pragma unroll
    for (int t = 0; t < 4; ++t) {
      v16h b = ((const v16h*)Wh)[(2 * s + g) * 64 + 16 * t + mn];
      acc[t] = __builtin_amdgcn_wmma_f32_16x16x32_f16(
          /*neg_a=*/false, a, /*neg_b=*/false, b,
          /*c_mod=*/(short)0, acc[t], /*reuse_a=*/false, /*reuse_b=*/false);
    }
  }
}

// ---------------- K3: th = (f16) ReLU(z @ W1 + b1) --------------------------
// One wave per 16-row strip; 4 N-tiles; 8 WMMAs per wave.
__global__ __launch_bounds__(128) void k_gemm1(const _Float16* __restrict__ zh,
                                               const _Float16* __restrict__ Wh1,
                                               const float* __restrict__ b1,
                                               _Float16* __restrict__ th,
                                               int nStrips) {
  int lane = threadIdx.x & 31;
  int g = lane >> 4, mn = lane & 15;
  int strip = blockIdx.x * 4 + (threadIdx.x >> 5);   // wave-uniform
  if (strip >= nStrips) return;                      // no barriers below: safe
  int R = strip * 16;
  v8f acc[4] = {};
  gemm_strip(zh, Wh1, R, g, mn, acc);
#pragma unroll
  for (int t = 0; t < 4; ++t) {
    int col = 16 * t + mn;
    float bias = b1[col];
#pragma unroll
    for (int d0 = 0; d0 < 8; ++d0) {                 // D elem d0 -> row d0+8g
      float v = acc[t][d0] + bias;
      v = v > 0.f ? v : 0.f;
      th[(size_t)(R + 8 * g + d0) * DIMF + col] = (_Float16)v;
    }
  }
}

// ---------------- K4: y2 = t @ W2 + b2 ; accumulate BN column sums ----------
__global__ __launch_bounds__(128) void k_gemm2(const _Float16* __restrict__ th,
                                               const _Float16* __restrict__ Wh2,
                                               const float* __restrict__ b2,
                                               float* __restrict__ y2,
                                               float* __restrict__ gsum,
                                               float* __restrict__ gsq,
                                               int nStrips) {
  __shared__ float s_sum[64];
  __shared__ float s_sq[64];
  int tid = threadIdx.x;
  if (tid < 64) { s_sum[tid] = 0.f; s_sq[tid] = 0.f; }
  __syncthreads();

  int lane = tid & 31;
  int g = lane >> 4, mn = lane & 15;
  int strip = blockIdx.x * 4 + (tid >> 5);
  if (strip < nStrips) {                             // wave-uniform guard
    int R = strip * 16;
    v8f acc[4] = {};
    gemm_strip(th, Wh2, R, g, mn, acc);
#pragma unroll
    for (int t = 0; t < 4; ++t) {
      int col = 16 * t + mn;
      float bias = b2[col];
      float cs = 0.f, cq = 0.f;
#pragma unroll
      for (int d0 = 0; d0 < 8; ++d0) {
        float v = acc[t][d0] + bias;
        y2[(size_t)(R + 8 * g + d0) * DIMF + col] = v;
        cs += v;
        cq += v * v;
      }
      atomicAdd(&s_sum[col], cs);                    // ds_add_f32
      atomicAdd(&s_sq[col],  cq);
    }
  }
  __syncthreads();
  if (tid < 64)        atomicAdd(&gsum[tid],      s_sum[tid]);
  else if (tid < 128)  atomicAdd(&gsq[tid - 64],  s_sq[tid - 64]);
}

// ---------------- K5: BN normalize + ReLU + residual ------------------------
__global__ void k_final(const float* __restrict__ h, const float* __restrict__ y2,
                        const float* __restrict__ gsum, const float* __restrict__ gsq,
                        const float* __restrict__ gamma, const float* __restrict__ beta,
                        float* __restrict__ out, int n, float invN) {
  int i = blockIdx.x * blockDim.x + threadIdx.x;
  if (i >= n) return;
  int c = i & 63;
  float mean = gsum[c] * invN;
  float var  = gsq[c] * invN - mean * mean;          // biased var (matches jnp.var)
  float rs   = rsqrtf(var + BN_EPS);
  float v = (y2[i] - mean) * rs * gamma[c] + beta[c];
  v = v > 0.f ? v : 0.f;
  out[i] = h[i] + v;
}

extern "C" void kernel_launch(void* const* d_in, const int* in_sizes, int n_in,
                              void* d_out, int out_size, void* d_ws, size_t ws_size,
                              hipStream_t stream) {
  const float* h     = (const float*)d_in[0];
  const float* W1    = (const float*)d_in[1];
  const float* b1    = (const float*)d_in[2];
  const float* W2    = (const float*)d_in[3];
  const float* b2    = (const float*)d_in[4];
  const float* gamma = (const float*)d_in[5];
  const float* beta  = (const float*)d_in[6];
  const int*   src   = (const int*)d_in[7];
  const int*   dst   = (const int*)d_in[8];

  int nNodes = in_sizes[0] / DIMF;                   // 50000
  int nEdges = in_sizes[7];                          // 800000
  size_t nd  = (size_t)nNodes * DIMF;

  // Workspace carving (all offsets 32B-aligned for nd = multiple of 64):
  //   [0,            4*nd)  : neigh f32  -> reused as y2 f32 after K2
  //   [4*nd,         6*nd)  : zh  f16
  //   [6*nd,         8*nd)  : th  f16
  //   [8*nd, 8*nd+16KB+..)  : Wh1(4096 h), Wh2(4096 h), stats(128 f)
  char* ws = (char*)d_ws;
  float*    neigh = (float*)ws;
  _Float16* zh    = (_Float16*)(ws + nd * 4);
  _Float16* th    = (_Float16*)(ws + nd * 6);
  _Float16* Wh1   = (_Float16*)(ws + nd * 8);
  _Float16* Wh2   = Wh1 + DIMF * DIMF;
  float*    stats = (float*)(Wh2 + DIMF * DIMF);
  float*    gsum  = stats;
  float*    gsq   = stats + 64;
  float*    y2    = neigh;                           // neigh dead after K2
  float*    out   = (float*)d_out;

  int nStrips = nNodes / 16;                         // 3125, exact

  k_zero<<<1024, 256, 0, stream>>>(neigh, stats, (int)(nd / 4));
  k_packW<<<(DIMF * DIMF + 255) / 256, 256, 0, stream>>>(W1, W2, Wh1, Wh2);
  {
    long long tot = (long long)nEdges * 32;
    k_scatter<<<(int)((tot + 255) / 256), 256, 0, stream>>>(h, src, dst, neigh, nEdges);
  }
  k_zh<<<(int)((nd + 255) / 256), 256, 0, stream>>>(h, neigh, zh, (int)nd);
  k_gemm1<<<(nStrips + 3) / 4, 128, 0, stream>>>(zh, Wh1, b1, th, nStrips);
  k_gemm2<<<(nStrips + 3) / 4, 128, 0, stream>>>(th, Wh2, b2, y2, gsum, gsq, nStrips);
  k_final<<<(int)((nd + 255) / 256), 256, 0, stream>>>(h, y2, gsum, gsq, gamma, beta,
                                                       out, (int)nd, 1.0f / (float)nNodes);
}
